// EffNetV2_4320737100289
// MI455X (gfx1250) — compile-verified, matrix operand order checked by reading, not run on hardware
//
#include <hip/hip_runtime.h>
#include <cstdint>

// ---------------------------------------------------------------------------
// MI455X (gfx1250) EffNetV2-SR. All GEMM work on v_wmma_f32_16x16x32_bf16.
// Weights are pre-packed per launch into WMMA B-fragment order so the hot
// loops issue only b128 loads + wmma. LDS tiles are channel-padded so A
// fragments are two ds_load_b128 each.
// ---------------------------------------------------------------------------

typedef __attribute__((ext_vector_type(16))) __bf16 v16bf;
typedef __attribute__((ext_vector_type(8)))  __bf16 v8bf;
typedef __attribute__((ext_vector_type(8)))  float  v8f;
typedef __attribute__((ext_vector_type(4)))  float  v4f;

#define DEV __device__ __forceinline__

DEV float leaky_f(float v) { return v >= 0.f ? v : 0.2f * v; }

// K index within a 32-wide chunk for 16-bit A/B WMMA fragments (ISA 7.12.2):
// lanes 0-15 hold K {0..7,16..23}; lanes 16-31 hold K {8..15,24..31}.
DEV int frag_k(int lane, int e) { return e + (e & 8) + ((lane & 16) ? 8 : 0); }

DEV v16bf join16(v8bf lo, v8bf hi) {
  return __builtin_shufflevector(lo, hi, 0, 1, 2, 3, 4, 5, 6, 7, 8, 9, 10, 11,
                                 12, 13, 14, 15);
}

static constexpr int H = 256, W = 256, B = 4, TW = 64;
static constexpr int GRID_TILES = B * H * (W / TW);   // 4096 row tiles

// ---------------------------------------------------------------------------
// Pack OIHW fp32 weights (taps = 9 for 3x3, 1 for 1x1) into bf16 WMMA B
// fragments: out[(((nt*KC+kc)*taps+tap)*32+lane)*16+e]. Zero-pads N/C.
// ---------------------------------------------------------------------------
__global__ __launch_bounds__(256) void pack_weights(
    const float* __restrict__ w, __bf16* __restrict__ out, int N, int C,
    int taps, int NT, int KC) {
  int idx = blockIdx.x * 256 + threadIdx.x;
  int total = NT * KC * taps * 512;
  if (idx >= total) return;
  int e = idx & 15;
  int lane = (idx >> 4) & 31;
  int frag = idx >> 9;             // (nt*KC + kc)*taps + tap
  int tap = frag % taps;
  int t2 = frag / taps;
  int kc = t2 % KC;
  int nt = t2 / KC;
  int n = nt * 16 + (lane & 15);
  int c = kc * 32 + frag_k(lane, e);
  float v = (n < N && c < C) ? w[(n * C + c) * taps + tap] : 0.f;
  out[idx] = (__bf16)v;
}

// ---------------------------------------------------------------------------
// Stem: 3x3 conv 1->24 + leaky (tiny; direct VALU)
// ---------------------------------------------------------------------------
__global__ __launch_bounds__(256) void stem_kernel(
    const float* __restrict__ x, const float* __restrict__ w,
    float* __restrict__ out) {
  int idx = blockIdx.x * 256 + threadIdx.x;
  int xc = idx & (W - 1);
  int y  = (idx >> 8) & (H - 1);
  int b  = idx >> 16;
  float patch[9];
#pragma unroll
  for (int dy = 0; dy < 3; ++dy)
#pragma unroll
    for (int dx = 0; dx < 3; ++dx) {
      int gy = y + dy - 1, gx = xc + dx - 1;
      patch[dy * 3 + dx] = (gy >= 0 && gy < H && gx >= 0 && gx < W)
                               ? x[((size_t)b * H + gy) * W + gx] : 0.f;
    }
#pragma unroll
  for (int c = 0; c < 24; ++c) {
    float a = 0.f;
#pragma unroll
    for (int t = 0; t < 9; ++t) a += patch[t] * w[c * 9 + t];
    out[(((size_t)b * 24 + c) * H + y) * W + xc] = leaky_f(a);
  }
}

// ---------------------------------------------------------------------------
// Fused non-SE MBConv: 3x3 conv (CIN->CHID) + leaky in LDS, 1x1 (CHID->COUT)
// + optional residual. One workgroup = 1 row x 64 cols, 8 wave32.
// ---------------------------------------------------------------------------
template <int CIN, int CHID, int COUT, bool IDENT>
__global__ __launch_bounds__(256) void mbconv_fused(
    const float* __restrict__ in, const __bf16* __restrict__ wpk_exp,
    const __bf16* __restrict__ wpk_proj, float* __restrict__ out) {
  constexpr int PW = TW + 2;
  constexpr int CINP  = ((CIN + 31) / 32) * 32;
  constexpr int CHIDP = ((CHID + 31) / 32) * 32;
  constexpr int KC1 = CINP / 32, NT1 = CHIDP / 16;
  constexpr int KC2 = CHIDP / 32, NT2 = (COUT + 15) / 16;

  __shared__ __align__(32) __bf16 sIn[3 * PW * CINP];
  __shared__ __align__(32) __bf16 sHid[TW * CHIDP];

  int wg = blockIdx.x;
  int x0 = (wg & 3) * TW;
  int y  = (wg >> 2) & (H - 1);
  int b  = wg >> 10;
  int tid  = threadIdx.x;
  int lane = tid & 31;
  int wave = tid >> 5;
  int m     = lane & 15;
  int khalf = (lane & 16) ? 8 : 0;

  const float* inB = in + (size_t)b * CIN * H * W;
  if (tid == 0) { __builtin_prefetch(wpk_exp, 0, 3); __builtin_prefetch(wpk_proj, 0, 3); }

  // Stage haloed input tile -> LDS bf16, channel-innermost, zero padded.
  for (int idx = tid; idx < 3 * PW * CINP; idx += 256) {
    int c = idx % CINP;
    int j = (idx / CINP) % PW;
    int r = idx / (CINP * PW);
    int gy = y + r - 1, gx = x0 + j - 1;
    float v = 0.f;
    if (c < CIN && gy >= 0 && gy < H && gx >= 0 && gx < W)
      v = inB[((size_t)c * H + gy) * W + gx];
    sIn[idx] = (__bf16)v;
  }
  __syncthreads();

  // Phase 1: 3x3 conv as 9 shifted GEMMs over packed-weight fragments.
  for (int nt = wave; nt < NT1; nt += 8) {
    v8f zero = {};
    v8f acc[4] = {zero, zero, zero, zero};
    for (int kc = 0; kc < KC1; ++kc) {
#pragma unroll
      for (int tap = 0; tap < 9; ++tap) {
        int dy = tap / 3, dx = tap % 3;
        const v8bf* bp = (const v8bf*)(wpk_exp +
            ((size_t)((nt * KC1 + kc) * 9 + tap) * 32 + lane) * 16);
        v16bf bf = join16(bp[0], bp[1]);
#pragma unroll
        for (int mt = 0; mt < 4; ++mt) {
          const __bf16* ap =
              &sIn[(dy * PW + mt * 16 + m + dx) * CINP + kc * 32 + khalf];
          v16bf af = join16(*(const v8bf*)ap, *(const v8bf*)(ap + 16));
          acc[mt] = __builtin_amdgcn_wmma_f32_16x16x32_bf16(
              false, af, false, bf, (short)0, acc[mt], false, false);
        }
      }
    }
    int ch = nt * 16 + m;
#pragma unroll
    for (int mt = 0; mt < 4; ++mt)
#pragma unroll
      for (int e = 0; e < 8; ++e) {
        int px = mt * 16 + e + khalf;
        float v = (ch < CHID) ? leaky_f(acc[mt][e]) : 0.f;  // zero the pad
        sHid[px * CHIDP + ch] = (__bf16)v;
      }
  }
  __syncthreads();

  // Phase 2: 1x1 projection + residual; waves take (mtile, ntile) pairs.
  float* outB = out + (size_t)b * COUT * H * W;
  for (int pair = wave; pair < 4 * NT2; pair += 8) {
    int mt = pair & 3;
    int nt = pair >> 2;
    v8f acc = {};
    for (int kc = 0; kc < KC2; ++kc) {
      const __bf16* ap = &sHid[(mt * 16 + m) * CHIDP + kc * 32 + khalf];
      v16bf af = join16(*(const v8bf*)ap, *(const v8bf*)(ap + 16));
      const v8bf* bp = (const v8bf*)(wpk_proj +
          ((size_t)(nt * KC2 + kc) * 32 + lane) * 16);
      v16bf bf = join16(bp[0], bp[1]);
      acc = __builtin_amdgcn_wmma_f32_16x16x32_bf16(
          false, af, false, bf, (short)0, acc, false, false);
    }
    int ch = nt * 16 + m;
    if (ch < COUT) {
      int px0 = mt * 16 + khalf;
      size_t base = ((size_t)ch * H + y) * W + x0 + px0;
      v4f lo = {acc[0], acc[1], acc[2], acc[3]};
      v4f hi = {acc[4], acc[5], acc[6], acc[7]};
      if (IDENT) {
        lo += *(const v4f*)&inB[base];
        hi += *(const v4f*)&inB[base + 4];
      }
      *(v4f*)&outB[base] = lo;
      *(v4f*)&outB[base + 4] = hi;
    }
  }
}

// ---------------------------------------------------------------------------
// SE kernel 1: 1x1 expand (CIN->CHID) + leaky -> bf16 hidden map (b128 out).
// ---------------------------------------------------------------------------
template <int CIN, int CHID>
__global__ __launch_bounds__(256) void se_expand(
    const float* __restrict__ in, const __bf16* __restrict__ wpk,
    __bf16* __restrict__ hid) {
  constexpr int CINP = ((CIN + 31) / 32) * 32;
  constexpr int KC = CINP / 32, NT = CHID / 16;
  __shared__ __align__(32) __bf16 sIn[TW * CINP];

  int wg = blockIdx.x;
  int x0 = (wg & 3) * TW;
  int y  = (wg >> 2) & (H - 1);
  int b  = wg >> 10;
  int tid  = threadIdx.x;
  int lane = tid & 31;
  int wave = tid >> 5;
  int m     = lane & 15;
  int khalf = (lane & 16) ? 8 : 0;

  const float* inB = in + (size_t)b * CIN * H * W;
  for (int idx = tid; idx < TW * CINP; idx += 256) {
    int c = idx % CINP;
    int px = idx / CINP;
    float v = (c < CIN) ? inB[((size_t)c * H + y) * W + x0 + px] : 0.f;
    sIn[idx] = (__bf16)v;
  }
  __syncthreads();

  for (int nt = wave; nt < NT; nt += 8) {
    v8f zero = {};
    v8f acc[4] = {zero, zero, zero, zero};
    for (int kc = 0; kc < KC; ++kc) {
      const v8bf* bp =
          (const v8bf*)(wpk + ((size_t)(nt * KC + kc) * 32 + lane) * 16);
      v16bf bf = join16(bp[0], bp[1]);
#pragma unroll
      for (int mt = 0; mt < 4; ++mt) {
        const __bf16* ap = &sIn[(mt * 16 + m) * CINP + kc * 32 + khalf];
        v16bf af = join16(*(const v8bf*)ap, *(const v8bf*)(ap + 16));
        acc[mt] = __builtin_amdgcn_wmma_f32_16x16x32_bf16(
            false, af, false, bf, (short)0, acc[mt], false, false);
      }
    }
    int ch = nt * 16 + m;
    size_t rowbase = ((size_t)(b * CHID + ch) * H + y) * W + x0;
#pragma unroll
    for (int mt = 0; mt < 4; ++mt) {
      v8bf pk;
#pragma unroll
      for (int e = 0; e < 8; ++e) pk[e] = (__bf16)leaky_f(acc[mt][e]);
      *(v8bf*)&hid[rowbase + mt * 16 + khalf] = pk;   // contiguous pixels
    }
  }
}

// ---------------------------------------------------------------------------
// SE kernel 2: depthwise 3x3 + leaky + per-(b,c) spatial-sum reduction.
// ---------------------------------------------------------------------------
template <int CHID>
__global__ __launch_bounds__(256) void se_dw(
    const __bf16* __restrict__ hid1, const float* __restrict__ wdw,
    __bf16* __restrict__ hid2, float* __restrict__ sums) {
  int wg = blockIdx.x;                  // b*CHID*H
  int y = wg & (H - 1);
  int c = (wg >> 8) % CHID;
  int b = wg / (CHID * H);
  int x = threadIdx.x;
  const __bf16* p = hid1 + (size_t)(b * CHID + c) * H * W;
  float a = 0.f;
#pragma unroll
  for (int dy = 0; dy < 3; ++dy) {
    int gy = y + dy - 1;
    if (gy < 0 || gy >= H) continue;
#pragma unroll
    for (int dx = 0; dx < 3; ++dx) {
      int gx = x + dx - 1;
      if (gx < 0 || gx >= W) continue;
      a += (float)p[gy * W + gx] * wdw[c * 9 + dy * 3 + dx];
    }
  }
  a = leaky_f(a);
  hid2[((size_t)(b * CHID + c) * H + y) * W + x] = (__bf16)a;

  __shared__ float red[256];
  red[threadIdx.x] = a;
  __syncthreads();
  for (int s = 128; s > 0; s >>= 1) {
    if (threadIdx.x < s) red[threadIdx.x] += red[threadIdx.x + s];
    __syncthreads();
  }
  if (threadIdx.x == 0) atomicAdd(&sums[b * CHID + c], red[0]);
}

// ---------------------------------------------------------------------------
// SE kernel 3: tiny MLP -> per-(b,c) sigmoid scales. One block per batch.
// ---------------------------------------------------------------------------
template <int CHID, int SEH>
__global__ __launch_bounds__(256) void se_scale_kernel(
    const float* __restrict__ sums, const float* __restrict__ fc1w,
    const float* __restrict__ fc1b, const float* __restrict__ fc2w,
    const float* __restrict__ fc2b, float* __restrict__ scale) {
  int b = blockIdx.x;
  int tid = threadIdx.x;
  __shared__ float t1[SEH];
  if (tid < SEH) {
    float a = fc1b[tid];
    for (int c = 0; c < CHID; ++c)
      a += (sums[b * CHID + c] * (1.f / (H * W))) * fc1w[tid * CHID + c];
    t1[tid] = a / (1.f + __expf(-a));   // silu
  }
  __syncthreads();
  for (int c = tid; c < CHID; c += 256) {
    float a = fc2b[c];
#pragma unroll
    for (int j = 0; j < SEH; ++j) a += t1[j] * fc2w[c * SEH + j];
    scale[b * CHID + c] = 1.f / (1.f + __expf(-a));
  }
}

// ---------------------------------------------------------------------------
// SE kernel 4: (hidden * scale) then 1x1 project (+ residual).
// ---------------------------------------------------------------------------
template <int CHID, int COUT, bool IDENT>
__global__ __launch_bounds__(256) void se_project(
    const __bf16* __restrict__ hid2, const float* __restrict__ scale,
    const __bf16* __restrict__ wpk, const float* __restrict__ res,
    float* __restrict__ out) {
  constexpr int KC = CHID / 32, NT = COUT / 16;
  __shared__ __align__(32) __bf16 sA[TW * CHID];

  int wg = blockIdx.x;
  int x0 = (wg & 3) * TW;
  int y  = (wg >> 2) & (H - 1);
  int b  = wg >> 10;
  int tid  = threadIdx.x;
  int lane = tid & 31;
  int wave = tid >> 5;
  int m     = lane & 15;
  int khalf = (lane & 16) ? 8 : 0;

  // Stage SE-scaled hidden tile (scale folded into the A operand).
  for (int idx = tid; idx < TW * CHID; idx += 256) {
    int c = idx % CHID;
    int px = idx / CHID;
    float v = (float)hid2[((size_t)(b * CHID + c) * H + y) * W + x0 + px] *
              scale[b * CHID + c];
    sA[px * CHID + c] = (__bf16)v;
  }
  __syncthreads();

  for (int pair = wave; pair < 4 * NT; pair += 8) {
    int mt = pair & 3;
    int nt = pair >> 2;
    v8f acc = {};
    for (int kc = 0; kc < KC; ++kc) {
      const __bf16* ap = &sA[(mt * 16 + m) * CHID + kc * 32 + khalf];
      v16bf af = join16(*(const v8bf*)ap, *(const v8bf*)(ap + 16));
      const v8bf* bp =
          (const v8bf*)(wpk + ((size_t)(nt * KC + kc) * 32 + lane) * 16);
      v16bf bf = join16(bp[0], bp[1]);
      acc = __builtin_amdgcn_wmma_f32_16x16x32_bf16(
          false, af, false, bf, (short)0, acc, false, false);
    }
    int ch = nt * 16 + m;
    int px0 = mt * 16 + khalf;
    size_t base = (((size_t)b * COUT + ch) * H + y) * W + x0 + px0;
    v4f lo = {acc[0], acc[1], acc[2], acc[3]};
    v4f hi = {acc[4], acc[5], acc[6], acc[7]};
    if (IDENT) {
      lo += *(const v4f*)&res[base];
      hi += *(const v4f*)&res[base + 4];
    }
    *(v4f*)&out[base] = lo;
    *(v4f*)&out[base + 4] = hi;
  }
}

// ---------------------------------------------------------------------------
// Final: double pixel-shuffle + 4x nearest upsample + scalar final conv.
// ---------------------------------------------------------------------------
__global__ __launch_bounds__(256) void final_kernel(
    const float* __restrict__ h, const float* __restrict__ x,
    const float* __restrict__ fw, float* __restrict__ out) {
  int idx = blockIdx.x * 256 + threadIdx.x;
  int X = idx & 1023;
  int Y = (idx >> 10) & 1023;
  int b = idx >> 20;
  int i = Y >> 2, j = X >> 2;
  int c16 = ((Y & 1) * 2 + (X & 1)) * 4 + ((Y >> 1) & 1) * 2 + ((X >> 1) & 1);
  float v = h[(((size_t)b * 16 + c16) * H + i) * W + j] +
            x[((size_t)b * H + i) * W + j];
  out[idx] = v * fw[0];
}

__global__ void zero_kernel(float* __restrict__ p, int n) {
  int i = blockIdx.x * 256 + threadIdx.x;
  if (i < n) p[i] = 0.f;
}

// ---------------------------------------------------------------------------
extern "C" void kernel_launch(void* const* d_in, const int* in_sizes, int n_in,
                              void* d_out, int out_size, void* d_ws,
                              size_t ws_size, hipStream_t stream) {
  auto f = [&](int i) { return (const float*)d_in[i]; };
  const float *x, *stem_w, *final_w;
  const float *wexp[7], *wproj[7];
  const float *wdw4, *fc1w4, *fc1b4, *fc2w4, *fc2b4;
  const float *wdw5, *fc1w5, *fc1b5, *fc2w5, *fc2b5;

  if (n_in > 0 && in_sizes[0] == 4 * 256 * 256) {
    // setup_inputs() insertion order.
    x = f(0); stem_w = f(1);
    wexp[0] = f(2);  wproj[0] = f(3);
    wexp[1] = f(4);  wproj[1] = f(5);
    wexp[2] = f(6);  wproj[2] = f(7);
    wexp[3] = f(8);  wproj[3] = f(9);
    wexp[4] = f(10); wdw4 = f(11); fc1w4 = f(12); fc1b4 = f(13);
    fc2w4 = f(14); fc2b4 = f(15); wproj[4] = f(16);
    wexp[5] = f(17); wdw5 = f(18); fc1w5 = f(19); fc1b5 = f(20);
    fc2w5 = f(21); fc2b5 = f(22); wproj[5] = f(23);
    wexp[6] = f(24); wproj[6] = f(25); final_w = f(26);
  } else {
    // jax sorted-pytree order: params{blocks,final_w,stem_w} then x.
    wexp[0] = f(0);  wproj[0] = f(1);
    wexp[1] = f(2);  wproj[1] = f(3);
    wexp[2] = f(4);  wproj[2] = f(5);
    wexp[3] = f(6);  wproj[3] = f(7);
    fc1b4 = f(8);  fc1w4 = f(9);  fc2b4 = f(10); fc2w4 = f(11);
    wdw4 = f(12); wexp[4] = f(13); wproj[4] = f(14);
    fc1b5 = f(15); fc1w5 = f(16); fc2b5 = f(17); fc2w5 = f(18);
    wdw5 = f(19); wexp[5] = f(20); wproj[5] = f(21);
    wexp[6] = f(22); wproj[6] = f(23); final_w = f(24); stem_w = f(25);
    x = f(26);
  }

  char* ws = (char*)d_ws;
  float*  bufA = (float*)ws;                           // 64 MB (<=64ch fp32)
  float*  bufB = (float*)(ws + (64ull << 20));         // 64 MB
  __bf16* hid1 = (__bf16*)(ws + (128ull << 20));       // 128 MB (256ch bf16)
  __bf16* hid2 = (__bf16*)(ws + (256ull << 20));       // 128 MB
  float*  sums = (float*)(ws + (384ull << 20));        // 4*256 f32
  float*  scal = sums + 4 * 256;                       // 4*256 f32
  __bf16* cur  = (__bf16*)(ws + (384ull << 20) + (1ull << 16));  // packed W

  dim3 blk(256);
  // ---- pack all conv weights into WMMA B-fragment order (bf16) ----
  auto packW = [&](const float* w, int N, int C, int taps, int NT,
                   int KC) -> const __bf16* {
    __bf16* dst = cur;
    int total = NT * KC * taps * 512;
    pack_weights<<<(total + 255) / 256, blk, 0, stream>>>(w, dst, N, C, taps,
                                                          NT, KC);
    cur += total;
    return dst;
  };
  const __bf16* pe0 = packW(wexp[0], 24, 24, 9, 2, 1);
  const __bf16* pp0 = packW(wproj[0], 24, 24, 1, 2, 1);
  const __bf16* pe1 = packW(wexp[1], 24, 24, 9, 2, 1);
  const __bf16* pp1 = packW(wproj[1], 24, 24, 1, 2, 1);
  const __bf16* pe2 = packW(wexp[2], 96, 24, 9, 6, 1);
  const __bf16* pp2 = packW(wproj[2], 48, 96, 1, 3, 3);
  const __bf16* pe3 = packW(wexp[3], 192, 48, 9, 12, 2);
  const __bf16* pp3 = packW(wproj[3], 48, 192, 1, 3, 6);
  const __bf16* pe4 = packW(wexp[4], 192, 48, 1, 12, 2);
  const __bf16* pp4 = packW(wproj[4], 64, 192, 1, 4, 6);
  const __bf16* pe5 = packW(wexp[5], 256, 64, 1, 16, 2);
  const __bf16* pp5 = packW(wproj[5], 64, 256, 1, 4, 8);
  const __bf16* pe6 = packW(wexp[6], 256, 64, 9, 16, 2);
  const __bf16* pp6 = packW(wproj[6], 16, 256, 1, 1, 8);

  stem_kernel<<<(B * H * W) / 256, blk, 0, stream>>>(x, stem_w, bufA);

  mbconv_fused<24, 24, 24, true ><<<GRID_TILES, blk, 0, stream>>>(bufA, pe0, pp0, bufB);
  mbconv_fused<24, 24, 24, true ><<<GRID_TILES, blk, 0, stream>>>(bufB, pe1, pp1, bufA);
  mbconv_fused<24, 96, 48, false><<<GRID_TILES, blk, 0, stream>>>(bufA, pe2, pp2, bufB);
  mbconv_fused<48, 192, 48, true><<<GRID_TILES, blk, 0, stream>>>(bufB, pe3, pp3, bufA);

  // Block 4 (SE): 48 -> hid 192 -> 64, no residual.
  zero_kernel<<<(4 * 192 + 255) / 256, blk, 0, stream>>>(sums, 4 * 192);
  se_expand<48, 192><<<GRID_TILES, blk, 0, stream>>>(bufA, pe4, hid1);
  se_dw<192><<<B * 192 * H, blk, 0, stream>>>(hid1, wdw4, hid2, sums);
  se_scale_kernel<192, 16><<<B, blk, 0, stream>>>(sums, fc1w4, fc1b4, fc2w4, fc2b4, scal);
  se_project<192, 64, false><<<GRID_TILES, blk, 0, stream>>>(hid2, scal, pp4, nullptr, bufB);

  // Block 5 (SE): 64 -> hid 256 -> 64, identity residual (bufB).
  zero_kernel<<<(4 * 256 + 255) / 256, blk, 0, stream>>>(sums, 4 * 256);
  se_expand<64, 256><<<GRID_TILES, blk, 0, stream>>>(bufB, pe5, hid1);
  se_dw<256><<<B * 256 * H, blk, 0, stream>>>(hid1, wdw5, hid2, sums);
  se_scale_kernel<256, 16><<<B, blk, 0, stream>>>(sums, fc1w5, fc1b5, fc2w5, fc2b5, scal);
  se_project<256, 64, true><<<GRID_TILES, blk, 0, stream>>>(hid2, scal, pp5, bufB, bufA);

  // Block 6: 64 -> hid 256 -> 16.
  mbconv_fused<64, 256, 16, false><<<GRID_TILES, blk, 0, stream>>>(bufA, pe6, pp6, bufB);

  final_kernel<<<(B * 1024 * 1024) / 256, blk, 0, stream>>>(bufB, x, final_w, (float*)d_out);
  (void)n_in; (void)out_size; (void)ws_size;
}
